// DGCNNTwoBranch_773094113294
// MI455X (gfx1250) — matrix operand order repaired and checked by reading, not compiled, not run on hardware
//
#include <hip/hip_runtime.h>
#include <hip/hip_bf16.h>
#include <stdint.h>

typedef __attribute__((ext_vector_type(16))) _Float16 v16h;
typedef __attribute__((ext_vector_type(8)))  float    v8f;

#define BDIM  16
#define NPTS  2048
#define KNN   20
#define NKTOT (NPTS * KNN)
#define XPAD  34   // LDS row pitch in halves (68B = 17 dwords, odd) -> conflict-poor

// ---------------------------------------------------------------------------
// kNN: one block per (b,n). Distances to LDS, then 20 iterative argmax passes.
// ---------------------------------------------------------------------------
__global__ void knn_kernel(const float* __restrict__ x, int C, int* __restrict__ idx)
{
    __shared__ float q[64];
    __shared__ float dist[NPTS];
    __shared__ float rv[256];
    __shared__ int   ri[256];
    const int b = blockIdx.x / NPTS;
    const int n = blockIdx.x % NPTS;
    const int t = threadIdx.x;
    for (int c = t; c < C; c += 256) q[c] = x[((size_t)b * C + c) * NPTS + n];
    __syncthreads();
    for (int m = t; m < NPTS; m += 256) {
        float acc = 0.f;
        for (int c = 0; c < C; ++c) {
            float d = x[((size_t)b * C + c) * NPTS + m] - q[c];
            acc += d * d;
        }
        dist[m] = -acc;                    // neg squared distance (ranking-equivalent)
    }
    __syncthreads();
    for (int kk = 0; kk < KNN; ++kk) {
        float best = -3.0e38f; int bi = 0x7fffffff;
        for (int m = t; m < NPTS; m += 256) {
            float v = dist[m];
            if (v > best) { best = v; bi = m; }
        }
        rv[t] = best; ri[t] = bi;
        __syncthreads();
        for (int s = 128; s > 0; s >>= 1) {
            if (t < s) {
                float vo = rv[t + s]; int io = ri[t + s];
                if (vo > rv[t] || (vo == rv[t] && io < ri[t])) { rv[t] = vo; ri[t] = io; }
            }
            __syncthreads();
        }
        if (t == 0) {
            int w = ri[0];
            idx[((size_t)b * NPTS + n) * KNN + kk] = w;
            dist[w] = -3.0e38f;
        }
        __syncthreads();
    }
}

// ---------------------------------------------------------------------------
// Edge feature gather: E[b][ch][n*K+k]; ch<C: center, ch>=C: neighbor-center
// ---------------------------------------------------------------------------
__global__ void edge_gather_kernel(const float* __restrict__ x, const int* __restrict__ idx,
                                   _Float16* __restrict__ E, int C)
{
    const size_t total = (size_t)BDIM * (2 * C) * NKTOT;
    size_t g = (size_t)blockIdx.x * blockDim.x + threadIdx.x;
    if (g >= total) return;
    int col = (int)(g % NKTOT);
    int ch  = (int)((g / NKTOT) % (2 * C));
    int b   = (int)(g / ((size_t)NKTOT * 2 * C));
    int n = col / KNN, k = col % KNN;
    float v;
    if (ch < C) {
        v = x[((size_t)b * C + ch) * NPTS + n];
    } else {
        int c = ch - C;
        int j = idx[((size_t)b * NPTS + n) * KNN + k];
        v = x[((size_t)b * C + c) * NPTS + j] - x[((size_t)b * C + c) * NPTS + n];
    }
    E[g] = (_Float16)v;
}

// ---------------------------------------------------------------------------
// Generic WMMA GEMM: OUT[b][o][col] = act( W[o][:] . IN[b][:][col] + bias[o] )
// W fp32 row-major (OxI, I even), IN f16 [b][I][cols], OUT f16/f32 [b][O][cols].
// Block = 256 thr = 8 waves; tile = 16 rows x 256 cols (32 cols/wave, 2 acc).
// K-step 32, double-buffered LDS, batched branchless staging, padded rows.
// cols must be a multiple of 256.
// ---------------------------------------------------------------------------
__global__ void __launch_bounds__(256)
gemm_wmma_kernel(const float* __restrict__ W, const float* __restrict__ bias,
                 const _Float16* __restrict__ IN,
                 _Float16* __restrict__ outH, float* __restrict__ outF,
                 int I, int O, int cols, int relu)
{
    __shared__ _Float16 sW[2][16][XPAD];
    __shared__ _Float16 sX[2][256][XPAD];
    const int t    = threadIdx.x;
    const int lane = t & 31;
    const int wv   = t >> 5;
    const int b    = blockIdx.z;
    const int ot   = blockIdx.y * 16;
    const int col0 = blockIdx.x * 256;

    const int mW  = t >> 4;            // weight-staging row   0..15
    const int kW  = (t & 15) * 2;      // weight-staging k     0,2,..,30
    const int oW  = ot + mW;
    const int oWc = (oW < O) ? oW : (O - 1);

    v8f acc0 = {0.f, 0.f, 0.f, 0.f, 0.f, 0.f, 0.f, 0.f};
    v8f acc1 = {0.f, 0.f, 0.f, 0.f, 0.f, 0.f, 0.f, 0.f};

    const int nsteps = (I + 31) >> 5;

    float w0r, w1r;          // staged weight pair (regs)
    uint4 xq[4];             // staged activation chunks (regs)

    // batched, branchless global->reg loads for one K-tile
    auto load_tile = [&](int k0) {
        int kg  = k0 + kW;
        int kgc = (kg < I - 1) ? kg : (I - 2);
        const float* wp = &W[(size_t)oWc * I + kgc];
        w0r = wp[0]; w1r = wp[1];
        #pragma unroll
        for (int it = 0; it < 4; ++it) {
            int s  = t + it * 256;
            int c8 = s & 31, kk = s >> 5;
            int kgx  = k0 + kk;
            int kgxc = (kgx < I) ? kgx : (I - 1);
            xq[it] = *(const uint4*)&IN[((size_t)b * I + kgxc) * cols + col0 + c8 * 8];
        }
    };
    // reg->LDS stores (transposed, zero-fill out-of-range k)
    auto store_tile = [&](int k0, int buf) {
        int kg = k0 + kW;
        bool okR = (oW < O);
        sW[buf][mW][kW]     = (okR && kg     < I) ? (_Float16)w0r : (_Float16)0.f;
        sW[buf][mW][kW + 1] = (okR && kg + 1 < I) ? (_Float16)w1r : (_Float16)0.f;
        #pragma unroll
        for (int it = 0; it < 4; ++it) {
            int s  = t + it * 256;
            int c8 = s & 31, kk = s >> 5;
            bool ok = (k0 + kk) < I;
            union { uint4 q; _Float16 h[8]; } u; u.q = xq[it];
            #pragma unroll
            for (int j = 0; j < 8; ++j)
                sX[buf][c8 * 8 + j][kk] = ok ? u.h[j] : (_Float16)0.f;
        }
    };

    load_tile(0);
    store_tile(0, 0);

    for (int st = 0; st < nsteps; ++st) {
        __syncthreads();
        const int buf = st & 1;
        if (st + 1 < nsteps) load_tile((st + 1) * 32);       // overlap with compute
        if (st + 2 < nsteps)
            __builtin_prefetch(&IN[((size_t)b * I + (size_t)(st + 2) * 32) * cols + col0 + t], 0, 0);

        // fragments in contiguous k order (k = 16*hi + s), same for A and B
        union { uint32_t u[8]; v16h v; } fa, fb0, fb1;
        const int hi = lane >> 4;
        const int m  = lane & 15;
        #pragma unroll
        for (int r = 0; r < 8; ++r) {
            int ka = 16 * hi + 2 * r;                                    // A layout (contiguous k, matches B)
            fa.u[r]  = *(const uint32_t*)&sW[buf][m][ka];
            int kb = 16 * hi + 2 * r;                                    // B layout
            fb0.u[r] = *(const uint32_t*)&sX[buf][wv * 32 + m][kb];
            fb1.u[r] = *(const uint32_t*)&sX[buf][wv * 32 + 16 + m][kb];
        }
        acc0 = __builtin_amdgcn_wmma_f32_16x16x32_f16(false, fa.v, false, fb0.v,
                                                      (short)0, acc0, false, false);
        acc1 = __builtin_amdgcn_wmma_f32_16x16x32_f16(false, fa.v, false, fb1.v,
                                                      (short)0, acc1, false, false);
        if (st + 1 < nsteps) store_tile((st + 1) * 32, buf ^ 1);
    }

    // epilogue: D element (row = r + 8*hi, col = lane&15), bias + ReLU
    const int hi   = lane >> 4;
    const int nn   = lane & 15;
    const int colA = col0 + wv * 32 + nn;
    const int colB = colA + 16;
    #pragma unroll
    for (int r = 0; r < 8; ++r) {
        int o = ot + r + 8 * hi;
        if (o < O) {
            float bv = bias[o];
            float va = acc0[r] + bv;
            float vb = acc1[r] + bv;
            if (relu) { va = fmaxf(va, 0.f); vb = fmaxf(vb, 0.f); }
            size_t rowo = ((size_t)b * O + o) * cols;
            if (outH) { outH[rowo + colA] = (_Float16)va; outH[rowo + colB] = (_Float16)vb; }
            else      { outF[rowo + colA] = va;           outF[rowo + colB] = vb; }
        }
    }
}

// ---------------------------------------------------------------------------
// Max over K (neighbor dim). Optional f16 out (into feat slab) and f32 out.
// ---------------------------------------------------------------------------
__global__ void maxk_kernel(const _Float16* __restrict__ in, int Ch,
                            _Float16* __restrict__ outH, int ohStride, int ohOff,
                            float* __restrict__ outF, int ofStride)
{
    size_t total = (size_t)BDIM * Ch * NPTS;
    size_t g = (size_t)blockIdx.x * blockDim.x + threadIdx.x;
    if (g >= total) return;
    int n  = (int)(g % NPTS);
    int ch = (int)((g / NPTS) % Ch);
    int b  = (int)(g / ((size_t)NPTS * Ch));
    const _Float16* p = in + ((size_t)b * Ch + ch) * NKTOT + (size_t)n * KNN;
    float m = (float)p[0];
    for (int k = 1; k < KNN; ++k) m = fmaxf(m, (float)p[k]);
    if (outH) outH[((size_t)b * ohStride + ohOff + ch) * NPTS + n] = (_Float16)m;
    if (outF) outF[((size_t)b * ofStride + ch) * NPTS + n] = m;
}

// Max over N (points) -> f32 global feature
__global__ void maxn_kernel(const _Float16* __restrict__ in, float* __restrict__ out, int Ch)
{
    __shared__ float r[256];
    int b = blockIdx.x / Ch, ch = blockIdx.x % Ch;
    int t = threadIdx.x;
    const _Float16* p = in + ((size_t)b * Ch + ch) * NPTS;
    float m = -3.0e38f;
    for (int n = t; n < NPTS; n += 256) m = fmaxf(m, (float)p[n]);
    r[t] = m; __syncthreads();
    for (int s = 128; s > 0; s >>= 1) { if (t < s) r[t] = fmaxf(r[t], r[t + s]); __syncthreads(); }
    if (t == 0) out[(size_t)b * Ch + ch] = r[0];
}

// ---------------------------------------------------------------------------
// T-Net FC head: 1024 -> 512 -> 256 -> 9, + identity -> 3x3 transform
// ---------------------------------------------------------------------------
__global__ void tnet_fc_kernel(const float* __restrict__ g,
                               const float* __restrict__ w0, const float* __restrict__ b0,
                               const float* __restrict__ w1, const float* __restrict__ b1,
                               const float* __restrict__ w2, const float* __restrict__ b2,
                               float* __restrict__ transWS, float* __restrict__ transOut)
{
    __shared__ float sg[1024];
    __shared__ float h0[512];
    __shared__ float h1[256];
    int b = blockIdx.x, t = threadIdx.x;
    for (int i = t; i < 1024; i += 256) sg[i] = g[(size_t)b * 1024 + i];
    __syncthreads();
    for (int o = t; o < 512; o += 256) {
        float a = b0[o];
        for (int i = 0; i < 1024; ++i) a += w0[(size_t)o * 1024 + i] * sg[i];
        h0[o] = fmaxf(a, 0.f);
    }
    __syncthreads();
    if (t < 256) {
        float a = b1[t];
        for (int i = 0; i < 512; ++i) a += w1[(size_t)t * 512 + i] * h0[i];
        h1[t] = fmaxf(a, 0.f);
    }
    __syncthreads();
    if (t < 9) {
        float a = b2[t];
        for (int i = 0; i < 256; ++i) a += w2[(size_t)t * 256 + i] * h1[i];
        if ((t % 4) == 0) a += 1.f;     // + eye(3)
        transWS[b * 9 + t]  = a;
        transOut[b * 9 + t] = a;
    }
}

// Apply 3x3 transform: xo[b][c][n] = sum_j trans[b][c][j] * x[b][j][n]
__global__ void transform_kernel(const float* __restrict__ x, const float* __restrict__ trans,
                                 float* __restrict__ xo)
{
    size_t total = (size_t)BDIM * 3 * NPTS;
    size_t g = (size_t)blockIdx.x * blockDim.x + threadIdx.x;
    if (g >= total) return;
    int n = (int)(g % NPTS);
    int c = (int)((g / NPTS) % 3);
    int b = (int)(g / (3 * NPTS));
    float a = 0.f;
    for (int j = 0; j < 3; ++j)
        a += trans[b * 9 + c * 3 + j] * x[((size_t)b * 3 + j) * NPTS + n];
    xo[g] = a;
}

// Global FC head: 1024 -> 256 -> 256 -> 128 -> 16 (last layer no ReLU)
__global__ void final_fc_kernel(const float* __restrict__ g,
                                const float* __restrict__ w0, const float* __restrict__ b0,
                                const float* __restrict__ w1, const float* __restrict__ b1,
                                const float* __restrict__ w2, const float* __restrict__ b2,
                                const float* __restrict__ w3, const float* __restrict__ b3,
                                float* __restrict__ out)
{
    __shared__ float sg[1024];
    __shared__ float h0[256];
    __shared__ float h1[256];
    __shared__ float h2[128];
    int b = blockIdx.x, t = threadIdx.x;
    for (int i = t; i < 1024; i += 256) sg[i] = g[(size_t)b * 1024 + i];
    __syncthreads();
    if (t < 256) { float a = b0[t]; for (int i = 0; i < 1024; ++i) a += w0[(size_t)t * 1024 + i] * sg[i]; h0[t] = fmaxf(a, 0.f); }
    __syncthreads();
    if (t < 256) { float a = b1[t]; for (int i = 0; i < 256;  ++i) a += w1[(size_t)t * 256 + i] * h0[i]; h1[t] = fmaxf(a, 0.f); }
    __syncthreads();
    if (t < 128) { float a = b2[t]; for (int i = 0; i < 256;  ++i) a += w2[(size_t)t * 256 + i] * h1[i]; h2[t] = fmaxf(a, 0.f); }
    __syncthreads();
    if (t < 16)  { float a = b3[t]; for (int i = 0; i < 128;  ++i) a += w3[(size_t)t * 128 + i] * h2[i]; out[b * 16 + t] = a; }
}

// Seg input: S[b][ch][n] = ch<1024 ? gfeat[b][ch] : feat[b][ch-1024][n]
__global__ void seg_build_kernel(const float* __restrict__ gfeat, const _Float16* __restrict__ feat,
                                 _Float16* __restrict__ S)
{
    size_t total = (size_t)BDIM * 1216 * NPTS;
    size_t g = (size_t)blockIdx.x * blockDim.x + threadIdx.x;
    if (g >= total) return;
    int n  = (int)(g % NPTS);
    int ch = (int)((g / NPTS) % 1216);
    int b  = (int)(g / ((size_t)1216 * NPTS));
    float v = (ch < 1024) ? gfeat[(size_t)b * 1024 + ch]
                          : (float)feat[((size_t)b * 192 + (ch - 1024)) * NPTS + n];
    S[g] = (_Float16)v;
}

// ---------------------------------------------------------------------------
extern "C" void kernel_launch(void* const* d_in, const int* in_sizes, int n_in,
                              void* d_out, int out_size, void* d_ws, size_t ws_size,
                              hipStream_t stream) {
    (void)in_sizes; (void)n_in; (void)out_size; (void)ws_size;
    const float* points   = (const float*)d_in[0];
    const float* tn_ec0_w = (const float*)d_in[1];  const float* tn_ec0_b = (const float*)d_in[2];
    const float* tn_ec1_w = (const float*)d_in[3];  const float* tn_ec1_b = (const float*)d_in[4];
    const float* tn_loc_w = (const float*)d_in[5];  const float* tn_loc_b = (const float*)d_in[6];
    const float* tn_g0_w  = (const float*)d_in[7];  const float* tn_g0_b  = (const float*)d_in[8];
    const float* tn_g1_w  = (const float*)d_in[9];  const float* tn_g1_b  = (const float*)d_in[10];
    const float* tn_lin_w = (const float*)d_in[11]; const float* tn_lin_b = (const float*)d_in[12];
    const float* ecw[3][2] = { { (const float*)d_in[13], (const float*)d_in[15] },
                               { (const float*)d_in[17], (const float*)d_in[19] },
                               { (const float*)d_in[21], (const float*)d_in[23] } };
    const float* ecb[3][2] = { { (const float*)d_in[14], (const float*)d_in[16] },
                               { (const float*)d_in[18], (const float*)d_in[20] },
                               { (const float*)d_in[22], (const float*)d_in[24] } };
    const float* loc_w  = (const float*)d_in[25]; const float* loc_b  = (const float*)d_in[26];
    const float* g0_w   = (const float*)d_in[27]; const float* g0_b   = (const float*)d_in[28];
    const float* g1_w   = (const float*)d_in[29]; const float* g1_b   = (const float*)d_in[30];
    const float* g2_w   = (const float*)d_in[31]; const float* g2_b   = (const float*)d_in[32];
    const float* go_w   = (const float*)d_in[33]; const float* go_b   = (const float*)d_in[34];
    const float* seg0_w = (const float*)d_in[35]; const float* seg0_b = (const float*)d_in[36];
    const float* seg1_w = (const float*)d_in[37]; const float* seg1_b = (const float*)d_in[38];
    const float* cs_w   = (const float*)d_in[39]; const float* cs_b   = (const float*)d_in[40];
    const float* mo_w   = (const float*)d_in[41]; const float* mo_b   = (const float*)d_in[42];

    float* out_global = (float*)d_out;                          // (B,16)
    float* out_mask   = (float*)d_out + 256;                    // (B,50,N)
    float* out_trans  = (float*)d_out + 256 + BDIM * 50 * NPTS; // (B,3,3)

    // ---- workspace arena (256B aligned regions) ----
    uint8_t* base = (uint8_t*)d_ws;
    size_t off = 0;
    auto arena = [&](size_t bytes) { size_t o = off; off += (bytes + 255) & ~(size_t)255; return o; };
    int*      idx    = (int*)     (base + arena((size_t)BDIM * NPTS * KNN * 4));
    _Float16* big0   = (_Float16*)(base + arena((size_t)BDIM * 128 * NKTOT * 2));
    _Float16* big1   = (_Float16*)(base + arena((size_t)BDIM * 128 * NKTOT * 2));
    _Float16* feat   = (_Float16*)(base + arena((size_t)BDIM * 192 * NPTS * 2));
    float*    x3     = (float*)   (base + arena((size_t)BDIM * 3 * NPTS * 4));
    float*    xcur   = (float*)   (base + arena((size_t)BDIM * 64 * NPTS * 4));
    float*    tnG    = (float*)   (base + arena((size_t)BDIM * 1024 * 4));
    float*    gfeat  = (float*)   (base + arena((size_t)BDIM * 1024 * 4));
    float*    transW = (float*)   (base + arena((size_t)BDIM * 9 * 4));

    auto gemm = [&](const float* W, const float* bias, const _Float16* IN,
                    _Float16* oH, float* oF, int I, int O, int cols, int relu) {
        dim3 grid(cols / 256, (O + 15) / 16, BDIM);
        gemm_wmma_kernel<<<grid, 256, 0, stream>>>(W, bias, IN, oH, oF, I, O, cols, relu);
    };
    auto ceil256 = [](size_t n) { return (unsigned)((n + 255) / 256); };

    // ================= T-Net branch =================
    knn_kernel<<<BDIM * NPTS, 256, 0, stream>>>(points, 3, idx);
    edge_gather_kernel<<<ceil256((size_t)BDIM * 6 * NKTOT), 256, 0, stream>>>(points, idx, big0, 3);
    gemm(tn_ec0_w, tn_ec0_b, big0, big1, nullptr, 6, 64, NKTOT, 1);
    gemm(tn_ec1_w, tn_ec1_b, big1, big0, nullptr, 64, 128, NKTOT, 1);
    maxk_kernel<<<ceil256((size_t)BDIM * 128 * NPTS), 256, 0, stream>>>(big0, 128, big1, 128, 0, nullptr, 0);
    gemm(tn_loc_w, tn_loc_b, big1, big0, nullptr, 128, 1024, NPTS, 1);
    maxn_kernel<<<BDIM * 1024, 256, 0, stream>>>(big0, tnG, 1024);
    tnet_fc_kernel<<<BDIM, 256, 0, stream>>>(tnG, tn_g0_w, tn_g0_b, tn_g1_w, tn_g1_b,
                                             tn_lin_w, tn_lin_b, transW, out_trans);
    transform_kernel<<<ceil256((size_t)BDIM * 3 * NPTS), 256, 0, stream>>>(points, transW, x3);

    // ================= Edge-conv blocks =================
    const float* xin = x3; int C = 3;
    for (int blk = 0; blk < 3; ++blk) {
        knn_kernel<<<BDIM * NPTS, 256, 0, stream>>>(xin, C, idx);
        edge_gather_kernel<<<ceil256((size_t)BDIM * 2 * C * NKTOT), 256, 0, stream>>>(xin, idx, big0, C);
        gemm(ecw[blk][0], ecb[blk][0], big0, big1, nullptr, 2 * C, 64, NKTOT, 1);
        gemm(ecw[blk][1], ecb[blk][1], big1, big0, nullptr, 64, 64, NKTOT, 1);
        maxk_kernel<<<ceil256((size_t)BDIM * 64 * NPTS), 256, 0, stream>>>(big0, 64, feat, 192, 64 * blk, xcur, 64);
        xin = xcur; C = 64;
    }

    // ================= Global head =================
    gemm(loc_w, loc_b, feat, big0, nullptr, 192, 1024, NPTS, 1);
    maxn_kernel<<<BDIM * 1024, 256, 0, stream>>>(big0, gfeat, 1024);
    final_fc_kernel<<<BDIM, 256, 0, stream>>>(gfeat, g0_w, g0_b, g1_w, g1_b,
                                              g2_w, g2_b, go_w, go_b, out_global);

    // ================= Segmentation head =================
    seg_build_kernel<<<ceil256((size_t)BDIM * 1216 * NPTS), 256, 0, stream>>>(gfeat, feat, big1);
    gemm(seg0_w, seg0_b, big1, big0, nullptr, 1216, 256, NPTS, 1);
    gemm(seg1_w, seg1_b, big0, big1, nullptr, 256, 256, NPTS, 1);
    gemm(cs_w,   cs_b,   big1, big0, nullptr, 256, 128, NPTS, 1);
    gemm(mo_w,   mo_b,   big0, nullptr, out_mask, 128, 50, NPTS, 0);  // no ReLU
}
